// SelfAttention_19774029431641
// MI455X (gfx1250) — matrix operand order repaired and checked by reading, not compiled, not run on hardware
//
#include <hip/hip_runtime.h>
#include <hip/hip_bf16.h>
#include <cstdint>

#define C_DIM   512
#define N_TOK   4096
#define BATCH   4
#define NGROUP  32
#define CPG     16   // channels per group

typedef __attribute__((ext_vector_type(16))) _Float16 hv16;
typedef __attribute__((ext_vector_type(8)))  _Float16 hv8;
typedef __attribute__((ext_vector_type(8)))  float    fv8;
typedef unsigned int u32;
typedef __attribute__((ext_vector_type(4))) u32 u32x4;
typedef __attribute__((ext_vector_type(8))) int i32x8;
typedef __attribute__((ext_vector_type(4))) int i32x4;

union Frag { hv16 v; hv8 h[2]; };

// ---- WMMA fragment loaders (CDNA5 16-bit A/B layouts, wave32) -------------
// A 16x32 (MxK): lanes 0-15 hold K=[0..7]+[16..23], lanes 16-31 K=[8..15]+[24..31]
__device__ inline hv16 load_frag_a(const _Float16* p, int lane) {
  const int h8 = (lane >> 4) << 3;
  Frag f;
  f.h[0] = *(const hv8*)(p + h8);
  f.h[1] = *(const hv8*)(p + 16 + h8);
  return f.v;
}
// B 32x16 (KxN): N = lane%16, lanes 0-15 K=[0..7]+[16..23], lanes 16-31 K=[8..15]+[24..31]
__device__ inline hv16 load_frag_b(const _Float16* p, int lane) {
  const int h8 = (lane >> 4) << 3;
  Frag f;
  f.h[0] = *(const hv8*)(p + h8);
  f.h[1] = *(const hv8*)(p + 16 + h8);
  return f.v;
}

#define WMMA_F16(a, b, c) \
  __builtin_amdgcn_wmma_f32_16x16x32_f16(false, (a), false, (b), (short)0, (c), false, false)

// ---- 4-lane group reductions via ds_swizzle (xor 1, xor 2) ----------------
template <int PAT>
__device__ inline float swz(float v) {
  return __int_as_float(__builtin_amdgcn_ds_swizzle(__float_as_int(v), PAT));
}
__device__ inline float qmax4(float v) {     // max across lanes {^1, ^2}
  v = fmaxf(v, swz<0x041F>(v));
  v = fmaxf(v, swz<0x081F>(v));
  return v;
}
__device__ inline float qsum4(float v) {
  v = v + swz<0x041F>(v);
  v = v + swz<0x081F>(v);
  return v;
}

// ---- Tensor Data Mover: 2-D tile global->LDS with LDS row padding ---------
__device__ inline void tdm_load_2d(u32 lds_off, const void* gsrc,
                                   u32 tensor_d0, u32 tensor_d1,
                                   u32 tile_d0, u32 tile_d1,
                                   unsigned long long d0_stride,
                                   u32 pad_interval, u32 pad_amount) {
  unsigned long long ga = (unsigned long long)gsrc;
  u32x4 g0;
  g0[0] = 1u;                                   // count=1, user descriptor
  g0[1] = lds_off;                              // lds_addr
  g0[2] = (u32)ga;                              // global_addr[31:0]
  g0[3] = (u32)((ga >> 32) & 0x01FFFFFFu) | (2u << 30);  // addr[56:32] | type=2
  i32x8 g1;
  g1[0] = (int)((1u << 16) /*data_size=2B*/ | (1u << 20) /*pad_en*/ |
                (pad_interval << 22) | (pad_amount << 25));
  g1[1] = (int)(tensor_d0 << 16);
  g1[2] = (int)((tensor_d0 >> 16) | (tensor_d1 << 16));
  g1[3] = (int)((tensor_d1 >> 16) | (tile_d0 << 16));
  g1[4] = (int)(tile_d1 & 0xFFFFu);             // tile_dim2 = 0
  g1[5] = (int)(u32)(d0_stride & 0xFFFFFFFFull);
  g1[6] = (int)(u32)((d0_stride >> 32) & 0xFFFFull);
  g1[7] = 0;
  i32x4 z4 = {};
#if defined(__clang_major__) && __clang_major__ >= 23
  i32x8 z8 = {};
  __builtin_amdgcn_tensor_load_to_lds(g0, g1, z4, z4, z8, 0);
#else
  __builtin_amdgcn_tensor_load_to_lds(g0, g1, z4, z4, 0);
#endif
}

// ---------------- GroupNorm statistics: one block per (b, group) -----------
__global__ void gn_stats_kernel(const float* __restrict__ x,
                                float* __restrict__ gmean, float* __restrict__ grstd) {
  const int bg = blockIdx.x;                    // b*32+g; group data is contiguous
  const float4* p4 = (const float4*)(x + (size_t)bg * (CPG * N_TOK));
  float s = 0.f, sq = 0.f;
  for (int i = threadIdx.x; i < (CPG * N_TOK) / 4; i += blockDim.x) {
    float4 v = p4[i];
    s  += v.x + v.y + v.z + v.w;
    sq += v.x*v.x + v.y*v.y + v.z*v.z + v.w*v.w;
  }
  __shared__ float rs[256], rq[256];
  rs[threadIdx.x] = s; rq[threadIdx.x] = sq;
  __syncthreads();
  for (int off = 128; off > 0; off >>= 1) {
    if ((int)threadIdx.x < off) { rs[threadIdx.x] += rs[threadIdx.x+off];
                                  rq[threadIdx.x] += rq[threadIdx.x+off]; }
    __syncthreads();
  }
  if (threadIdx.x == 0) {
    const float inv = 1.0f / (float)(CPG * N_TOK);
    float m = rs[0] * inv;
    float var = rq[0] * inv - m * m;
    gmean[bg] = m;
    grstd[bg] = rsqrtf(var + 1e-6f);
  }
}

// ---------------- fp32 -> f16 weight conversion ----------------------------
__global__ void cvt_w_kernel(const float* __restrict__ a, const float* __restrict__ b,
                             const float* __restrict__ c, const float* __restrict__ d,
                             _Float16* __restrict__ A, _Float16* __restrict__ B,
                             _Float16* __restrict__ Cc, _Float16* __restrict__ D) {
  int i = blockIdx.x * blockDim.x + threadIdx.x;
  if (i < C_DIM * C_DIM) {
    A[i]  = (_Float16)a[i];  B[i] = (_Float16)b[i];
    Cc[i] = (_Float16)c[i];  D[i] = (_Float16)d[i];
  }
}

// ------- GN apply + transpose [B,C,N] fp32 -> h16 [B,N,C] f16 --------------
__global__ void gn_apply_kernel(const float* __restrict__ x,
                                const float* __restrict__ gmean, const float* __restrict__ grstd,
                                const float* __restrict__ w, const float* __restrict__ bias,
                                _Float16* __restrict__ h16) {
  const int b = blockIdx.z, c0 = blockIdx.y * 32, n0 = blockIdx.x * 32;
  __shared__ float tile[32][33];
  const int t = threadIdx.x;
#pragma unroll
  for (int p = 0; p < 4; ++p) {
    int cr = p * 8 + (t >> 5), nr = t & 31;
    int c = c0 + cr;
    float v = x[(size_t)(b * C_DIM + c) * N_TOK + n0 + nr];
    int g = b * NGROUP + (c >> 4);
    tile[cr][nr] = (v - gmean[g]) * grstd[g] * w[c] + bias[c];
  }
  __syncthreads();
#pragma unroll
  for (int p = 0; p < 4; ++p) {
    int nr = p * 8 + (t >> 5), cr = t & 31;
    h16[(size_t)(b * N_TOK + n0 + nr) * C_DIM + c0 + cr] = (_Float16)tile[cr][nr];
  }
}

// ------- QKV: three fused GEMMs, shared A fragments; V stored transposed ---
__global__ void __launch_bounds__(256)
qkv_kernel(const _Float16* __restrict__ h16,
           const _Float16* __restrict__ Wq, const _Float16* __restrict__ Wk,
           const _Float16* __restrict__ Wv,
           const float* __restrict__ bq, const float* __restrict__ bk,
           const float* __restrict__ bv,
           _Float16* __restrict__ Q, _Float16* __restrict__ K,
           _Float16* __restrict__ Vt) {
  const int lane = threadIdx.x & 31, wid = threadIdx.x >> 5;
  const int tile = blockIdx.x * 8 + wid;
  const int b  = tile / ((N_TOK/16) * (C_DIM/16));
  const int rem = tile % ((N_TOK/16) * (C_DIM/16));
  const int n0 = (rem / (C_DIM/16)) * 16;
  const int c0 = (rem % (C_DIM/16)) * 16;

  fv8 aq = {}, ak = {}, av = {};
  const _Float16* arow = h16 + (size_t)(b * N_TOK + n0 + (lane & 15)) * C_DIM;
  const _Float16* qrow = Wq + (size_t)(c0 + (lane & 15)) * C_DIM;
  const _Float16* krow = Wk + (size_t)(c0 + (lane & 15)) * C_DIM;
  const _Float16* vrow = Wv + (size_t)(c0 + (lane & 15)) * C_DIM;
#pragma unroll 4
  for (int kk = 0; kk < C_DIM; kk += 32) {
    hv16 a = load_frag_a(arow + kk, lane);
    aq = WMMA_F16(a, load_frag_b(qrow + kk, lane), aq);
    ak = WMMA_F16(a, load_frag_b(krow + kk, lane), ak);
    av = WMMA_F16(a, load_frag_b(vrow + kk, lane), av);
  }
  const int col = lane & 15, h8 = (lane >> 4) * 8;
  const float biq = bq[c0+col], bik = bk[c0+col], biv = bv[c0+col];
#pragma unroll
  for (int r = 0; r < 8; ++r) {
    size_t idx = (size_t)(b * N_TOK + n0 + h8 + r) * C_DIM + c0 + col;
    Q[idx] = (_Float16)(aq[r] + biq);
    K[idx] = (_Float16)(ak[r] + bik);
  }
  hv8 vp;                                      // lane's 8 rows are contiguous n in Vt
#pragma unroll
  for (int r = 0; r < 8; ++r) vp[r] = (_Float16)(av[r] + biv);
  *(hv8*)(Vt + (size_t)(b * C_DIM + c0 + col) * N_TOK + n0 + h8) = vp;
}

// ------- Flash attention: 64 queries/block, 32-key tiles, TDM ping-pong ----
#define BK 32
#define NB (N_TOK / BK)

__global__ void __launch_bounds__(256)
flash_kernel(const _Float16* __restrict__ Q, const _Float16* __restrict__ K,
             const _Float16* __restrict__ Vt, _Float16* __restrict__ O) {
  __shared__ __align__(16) _Float16 Ks2[2][BK][520];    // 66,560 B (TDM row pad 16B)
  __shared__ __align__(16) _Float16 Vts2[2][C_DIM][40]; // 81,920 B (TDM row pad 16B)
  __shared__ __align__(16) float    Ss[64][34];         //  8,704 B
  __shared__ __align__(16) _Float16 Ps[64][40];         //  5,120 B
  __shared__ float m_state[64], l_state[64], alpha_s[64]; // 768 B  => 163,072 B total

  const int t = threadIdx.x, lane = t & 31, wid = t >> 5;
  const int b = blockIdx.y, q0 = blockIdx.x * 64;
  const float scale = 0.044194173824159216f;   // 1/sqrt(512)

  if (t < 64) { m_state[t] = -1e30f; l_state[t] = 0.f; }

  const fv8 fz = {};
  fv8 accO[4][4];                              // 64 rows x 64 cols per wave
#pragma unroll
  for (int i = 0; i < 4; ++i)
#pragma unroll
    for (int j = 0; j < 4; ++j) accO[i][j] = fz;

  const int rt = wid >> 1, ct = wid & 1;       // S phase: 64x32 = 8 subtiles
  const _Float16* qrowA = Q + (size_t)(b * N_TOK + q0 + rt*16 + (lane & 15)) * C_DIM;
  const _Float16* kbase = K  + (size_t)(b * N_TOK) * C_DIM;
  const _Float16* vbase = Vt + (size_t)(b * C_DIM) * N_TOK;

  // prologue: fill buffer 0
  if (t < 32) {
    tdm_load_2d((u32)(uintptr_t)&Ks2[0][0][0], kbase,
                512, 4096, 512, BK, 512ull, 7u, 3u);
    tdm_load_2d((u32)(uintptr_t)&Vts2[0][0][0], vbase,
                4096, 512, BK, 512, 4096ull, 3u, 3u);
  }

  for (int kb = 0; kb < NB; ++kb) {
    const int cur = kb & 1;
    if (t < 32) __builtin_amdgcn_s_wait_tensorcnt(0);   // buf[cur] landed
    __syncthreads();
    if (t < 32 && kb + 1 < NB) {                        // prefetch buf[cur^1]
      const int k1 = (kb + 1) * BK;
      tdm_load_2d((u32)(uintptr_t)&Ks2[cur ^ 1][0][0], kbase + (size_t)k1 * C_DIM,
                  512, 4096, 512, BK, 512ull, 7u, 3u);
      tdm_load_2d((u32)(uintptr_t)&Vts2[cur ^ 1][0][0], vbase + k1,
                  4096, 512, BK, 512, 4096ull, 3u, 3u);
    }

    // Phase A: S = scale * Q Kb^T (each wave one 16x16 subtile, K=512)
    fv8 s = fz;
#pragma unroll 4
    for (int kk = 0; kk < C_DIM; kk += 32) {
      hv16 a   = load_frag_a(qrowA + kk, lane);
      hv16 bfr = load_frag_b(&Ks2[cur][ct*16 + (lane & 15)][kk], lane);
      s = WMMA_F16(a, bfr, s);
    }
    {
      const int h8 = (lane >> 4) * 8, col = ct*16 + (lane & 15);
#pragma unroll
      for (int r = 0; r < 8; ++r) Ss[rt*16 + h8 + r][col] = s[r] * scale;
    }
    __syncthreads();

    // Phase B: online softmax — 4 lanes per row, ds_swizzle reductions
    {
      const int row = t >> 2, seg = t & 3;               // 8 cols per thread
      float lm = -1e30f;
#pragma unroll
      for (int i2 = 0; i2 < 8; ++i2) lm = fmaxf(lm, Ss[row][seg*8 + i2]);
      lm = qmax4(lm);
      const float mo = m_state[row];
      const float mn = fmaxf(mo, lm);
      const float al = __expf(mo - mn);
      float ps = 0.f;
#pragma unroll
      for (int i2 = 0; i2 < 8; ++i2) {
        float p = __expf(Ss[row][seg*8 + i2] - mn);
        Ps[row][seg*8 + i2] = (_Float16)p;
        ps += p;
      }
      ps = qsum4(ps);
      if (seg == 0) {
        alpha_s[row] = al;
        m_state[row] = mn;
        l_state[row] = l_state[row] * al + ps;
      }
    }
    __syncthreads();

    // Phase C: O = alpha*O + P @ Vb (wave owns 64 rows x 64 cols, K=32)
    const int h8 = (lane >> 4) * 8;
#pragma unroll
    for (int i = 0; i < 4; ++i) {
      float al[8];
#pragma unroll
      for (int r = 0; r < 8; ++r) al[r] = alpha_s[i*16 + h8 + r];
#pragma unroll
      for (int j = 0; j < 4; ++j)
#pragma unroll
        for (int r = 0; r < 8; ++r) accO[i][j][r] *= al[r];
      hv16 a = load_frag_a(&Ps[i*16 + (lane & 15)][0], lane);
#pragma unroll
      for (int j = 0; j < 4; ++j) {
        hv16 bfr = load_frag_b(&Vts2[cur][wid*64 + j*16 + (lane & 15)][0], lane);
        accO[i][j] = WMMA_F16(a, bfr, accO[i][j]);
      }
    }
  }
  __syncthreads();

  // finalize: O /= l
  const int h8 = (lane >> 4) * 8, col = lane & 15;
#pragma unroll
  for (int i = 0; i < 4; ++i) {
    float inv[8];
#pragma unroll
    for (int r = 0; r < 8; ++r) inv[r] = 1.0f / l_state[i*16 + h8 + r];
#pragma unroll
    for (int j = 0; j < 4; ++j)
#pragma unroll
      for (int r = 0; r < 8; ++r)
        O[(size_t)(b * N_TOK + q0 + i*16 + h8 + r) * C_DIM + wid*64 + j*16 + col] =
            (_Float16)(accO[i][j][r] * inv[r]);
  }
}

// ------- out = O @ Wo^T + bo, transpose, residual add, fp32 out ------------
__global__ void __launch_bounds__(256)
outproj_kernel(const _Float16* __restrict__ O, const _Float16* __restrict__ Wo,
               const float* __restrict__ bo, const float* __restrict__ x,
               float* __restrict__ y) {
  __shared__ float tr[8][16][17];
  const int lane = threadIdx.x & 31, wid = threadIdx.x >> 5;
  const int tile = blockIdx.x * 8 + wid;
  const int b  = tile / ((N_TOK/16) * (C_DIM/16));
  const int rem = tile % ((N_TOK/16) * (C_DIM/16));
  const int n0 = (rem / (C_DIM/16)) * 16;
  const int c0 = (rem % (C_DIM/16)) * 16;

  fv8 acc = {};
  const _Float16* arow = O  + (size_t)(b * N_TOK + n0 + (lane & 15)) * C_DIM;
  const _Float16* brow = Wo + (size_t)(c0 + (lane & 15)) * C_DIM;
#pragma unroll 4
  for (int kk = 0; kk < C_DIM; kk += 32)
    acc = WMMA_F16(load_frag_a(arow + kk, lane), load_frag_b(brow + kk, lane), acc);

  const int col = lane & 15, h8 = (lane >> 4) * 8;
  const float bias = bo[c0 + col];
#pragma unroll
  for (int r = 0; r < 8; ++r) tr[wid][h8 + r][col] = acc[r] + bias;
  __syncthreads();
#pragma unroll
  for (int i2 = 0; i2 < 8; ++i2) {
    int cr = h8 + i2;
    size_t idx = (size_t)(b * C_DIM + c0 + cr) * N_TOK + n0 + col;
    y[idx] = x[idx] + tr[wid][col][cr];      // coalesced along W, residual add
  }
}

// ---------------------------------------------------------------------------
extern "C" void kernel_launch(void* const* d_in, const int* in_sizes, int n_in,
                              void* d_out, int out_size, void* d_ws, size_t ws_size,
                              hipStream_t stream) {
  (void)in_sizes; (void)n_in; (void)out_size; (void)ws_size;
  const float* x   = (const float*)d_in[0];
  const float* gnw = (const float*)d_in[1];
  const float* gnb = (const float*)d_in[2];
  const float* Wq  = (const float*)d_in[3];
  const float* bq  = (const float*)d_in[4];
  const float* Wk  = (const float*)d_in[5];
  const float* bk  = (const float*)d_in[6];
  const float* Wv  = (const float*)d_in[7];
  const float* bv  = (const float*)d_in[8];
  const float* Wo  = (const float*)d_in[9];
  const float* bo  = (const float*)d_in[10];
  float* y = (float*)d_out;

  char* ws = (char*)d_ws;
  size_t off = 0;
  auto alloc = [&](size_t bytes) {
    char* p = ws + off;
    off += bytes;
    off = (off + 255) & ~(size_t)255;
    return p;
  };
  const size_t BNC = (size_t)BATCH * N_TOK * C_DIM;
  float*    gmean = (float*)alloc(BATCH * NGROUP * 4);
  float*    grstd = (float*)alloc(BATCH * NGROUP * 4);
  _Float16* h16   = (_Float16*)alloc(BNC * 2);
  _Float16* Wq16  = (_Float16*)alloc((size_t)C_DIM * C_DIM * 2);
  _Float16* Wk16  = (_Float16*)alloc((size_t)C_DIM * C_DIM * 2);
  _Float16* Wv16  = (_Float16*)alloc((size_t)C_DIM * C_DIM * 2);
  _Float16* Wo16  = (_Float16*)alloc((size_t)C_DIM * C_DIM * 2);
  _Float16* Q16   = (_Float16*)alloc(BNC * 2);
  _Float16* K16   = (_Float16*)alloc(BNC * 2);
  _Float16* Vt16  = (_Float16*)alloc(BNC * 2);
  _Float16* O16   = (_Float16*)alloc(BNC * 2);

  gn_stats_kernel<<<BATCH * NGROUP, 256, 0, stream>>>(x, gmean, grstd);
  cvt_w_kernel<<<(C_DIM * C_DIM + 255) / 256, 256, 0, stream>>>(
      Wq, Wk, Wv, Wo, Wq16, Wk16, Wv16, Wo16);
  gn_apply_kernel<<<dim3(N_TOK / 32, C_DIM / 32, BATCH), 256, 0, stream>>>(
      x, gmean, grstd, gnw, gnb, h16);
  qkv_kernel<<<(BATCH * (N_TOK / 16) * (C_DIM / 16)) / 8, 256, 0, stream>>>(
      h16, Wq16, Wk16, Wv16, bq, bk, bv, Q16, K16, Vt16);
  flash_kernel<<<dim3(N_TOK / 64, BATCH), 256, 0, stream>>>(Q16, K16, Vt16, O16);
  outproj_kernel<<<(BATCH * (N_TOK / 16) * (C_DIM / 16)) / 8, 256, 0, stream>>>(
      O16, Wo16, bo, x, y);
}
